// MultiModeEncoder_35519379538330
// MI455X (gfx1250) — matrix operand order repaired and checked by reading, not compiled, not run on hardware
//
#include <hip/hip_runtime.h>

// ---------------------------------------------------------------------------
// MultiModeEncoder for MI455X (gfx1250, wave32).
//
// Roofline strategy: the 3 big edge GEMMs/layer ((294912,128)x(128,256))
// dominate FLOPs (~58 GFLOP/layer). edge/node features are layer-invariant,
// so we convert them to bf16 ONCE (halves read traffic, enables
// v_wmma_f32_16x16x32_bf16, the fast CDNA5 matrix path); all accumulation /
// LN / softmax / segment-reduction stays f32.
//
// This revision stages each 128-row block's B panel (contiguous thanks to the
// pre-pack layout) into LDS with gfx1250 async copies
// (global_load_async_to_lds_b128 + s_wait_asynccnt + barrier), then feeds
// WMMA from ds_load_b128 — 8x less global B traffic and shorter wait chains.
// Workspace use: ~490 MB.
// ---------------------------------------------------------------------------

#define E_EDGE   49152
#define N_EGO_   1024
#define M_MODES_ 6
#define HID_     256
#define HH_      128
#define HEADS_   8
#define L_       3
#define R_EDGE   (E_EDGE * M_MODES_)   /* 294912 rows for edge GEMMs */
#define R_EGO    (N_EGO_ * M_MODES_)   /* 6144  rows for ego  GEMMs */

typedef __attribute__((ext_vector_type(16))) __bf16  v16bf;
typedef __attribute__((ext_vector_type(8)))  float   v8f;

union Frag { uint4 q[2]; v16bf v; };

__device__ __forceinline__ unsigned short f2bf(float f) {
  union { float f; unsigned u; } x; x.f = f;
  unsigned r = x.u + 0x7FFFu + ((x.u >> 16) & 1u);   // round-to-nearest-even
  return (unsigned short)(r >> 16);
}
__device__ __forceinline__ float bf2f(unsigned short h) {
  union { unsigned u; float f; } x; x.u = ((unsigned)h) << 16; return x.f;
}

// ---------------------------------------------------------------------------
// f32 -> bf16 bulk convert (grid-stride)
// ---------------------------------------------------------------------------
__global__ void mme_cvt_bf16(const float* __restrict__ src,
                             unsigned short* __restrict__ dst, size_t n) {
  size_t i = (size_t)blockIdx.x * blockDim.x + threadIdx.x;
  size_t stride = (size_t)gridDim.x * blockDim.x;
  for (; i < n; i += stride) dst[i] = f2bf(src[i]);
}

// ---------------------------------------------------------------------------
// Pack a row-major f32 weight [K,N] into bf16 WMMA B fragments.
// Per (kt, nt) 32x16 tile: lane l holds column n = nt*16 + (l&15),
// K run kb = kt*32 + (l>>4)*16, 16 contiguous bf16 per lane. A whole
// 64-column group's panel is contiguous => async-stageable as a flat copy.
// ---------------------------------------------------------------------------
__global__ void mme_pack_b(const float* __restrict__ B,
                           unsigned short* __restrict__ P, int K, int N) {
  int lane = threadIdx.x;               // blockDim.x == 32
  int ktn  = K >> 5;
  int kt   = blockIdx.x % ktn;
  int nt   = blockIdx.x / ktn;
  int n    = nt * 16 + (lane & 15);
  int kb   = kt * 32 + (lane >> 4) * 16;
  unsigned short* dst = P + ((size_t)(nt * ktn + kt) * 32 + lane) * 16;
#pragma unroll
  for (int j = 0; j < 16; ++j)
    dst[j] = f2bf(B[(size_t)(kb + j) * N + n]);
}

// ---------------------------------------------------------------------------
// LayerNorm over last dim (256), f32 in -> bf16 out. One wave per row.
// ---------------------------------------------------------------------------
__global__ void mme_layernorm(const float* __restrict__ x,
                              const float* __restrict__ w,
                              const float* __restrict__ b,
                              unsigned short* __restrict__ out, int rows) {
  int lane = threadIdx.x & 31;
  int row  = blockIdx.x * (blockDim.x >> 5) + (threadIdx.x >> 5);
  if (row >= rows) return;
  const float* xp = x + (size_t)row * HID_ + lane * 8;
  float vals[8]; float s = 0.f, s2 = 0.f;
#pragma unroll
  for (int j = 0; j < 8; ++j) { vals[j] = xp[j]; s += vals[j]; s2 += vals[j]*vals[j]; }
#pragma unroll
  for (int off = 16; off; off >>= 1) {
    s  += __shfl_xor(s,  off);
    s2 += __shfl_xor(s2, off);
  }
  float mu  = s  * (1.f / HID_);
  float var = s2 * (1.f / HID_) - mu * mu;
  float rs  = rsqrtf(var + 1e-5f);
  unsigned short* op = out + (size_t)row * HID_ + lane * 8;
#pragma unroll
  for (int j = 0; j < 8; ++j) {
    int ch = lane * 8 + j;
    op[j] = f2bf((vals[j] - mu) * rs * w[ch] + b[ch]);
  }
}

// ---------------------------------------------------------------------------
// bf16 WMMA GEMM: C[R,N] = A1@B1 (+ A2@B2) + bias1 (+bias2), optional ReLU,
// optional f32 residual, output f32 or bf16.
//
// Block = 256 threads = 8 waves = 128 rows x 64 cols. The block's B panel(s)
// (32 KB max) are staged into LDS with gfx1250 async global->LDS copies, then
// each wave runs the K loop: A fragments straight from row-major bf16
// (two 16B chunks/lane holding k = 16*half + 0..15, same order as packed B), B from LDS.
// ---------------------------------------------------------------------------
__global__ void __launch_bounds__(256) mme_gemm_wmma(
    const unsigned short* __restrict__ A1, const unsigned short* __restrict__ B1,
    const unsigned short* __restrict__ A2, const unsigned short* __restrict__ B2,
    const float* __restrict__ bias1, const float* __restrict__ bias2,
    const float* __restrict__ resid,
    float* __restrict__ outF, unsigned short* __restrict__ outB,
    int R, int K, int N, int relu) {
  __shared__ unsigned short lds_b[16384];            // 32 KB panel storage
  const int tid  = threadIdx.x;
  const int lane = tid & 31;
  const int wv   = tid >> 5;
  const int rbn  = R >> 7;                           // 128-row blocks
  const int rb   = (int)blockIdx.x % rbn;
  const int cg   = (int)blockIdx.x / rbn;            // 64-wide column group
  const int ktn  = K >> 5;
  const int half = lane >> 4;
  const int l15  = lane & 15;

  // ---- async-stage B panel(s): contiguous panelElems ushorts per panel ----
  const int panelElems = 4 * ktn * 32 * 16;          // <= 16384
  {
    const unsigned short* gB1 = B1 + (size_t)(cg * 4) * ktn * 32 * 16;
    const int nchunks = panelElems >> 3;             // 16B chunks
    for (int i = tid; i < nchunks; i += 256) {
      unsigned loff = (unsigned)(unsigned long long)&lds_b[i * 8];
      asm volatile("global_load_async_to_lds_b128 %0, %1, off"
                   :: "v"(loff), "v"(gB1 + (size_t)i * 8) : "memory");
    }
    if (A2) {
      const unsigned short* gB2 = B2 + (size_t)(cg * 4) * ktn * 32 * 16;
      for (int i = tid; i < nchunks; i += 256) {
        unsigned loff = (unsigned)(unsigned long long)&lds_b[panelElems + i * 8];
        asm volatile("global_load_async_to_lds_b128 %0, %1, off"
                     :: "v"(loff), "v"(gB2 + (size_t)i * 8) : "memory");
      }
    }
    asm volatile("s_wait_asynccnt 0x0" ::: "memory");
  }
  __syncthreads();

  v8f acc[4];
#pragma unroll
  for (int t = 0; t < 4; ++t) acc[t] = (v8f)0.0f;

  const int rt   = rb * 8 + wv;
  const int arow = rt * 16 + l15;
  const unsigned short* a1p = A1 + (size_t)arow * K + half * 16;
  const unsigned short* a2p = A2 ? (A2 + (size_t)arow * K + half * 16)
                                 : (const unsigned short*)0;

  for (int kt = 0; kt < ktn; ++kt) {
    Frag a;
    a.q[0] = *(const uint4*)(a1p + kt * 32);
    a.q[1] = *(const uint4*)(a1p + kt * 32 + 8);
#pragma unroll
    for (int t = 0; t < 4; ++t) {
      Frag b;
      const unsigned short* bp = &lds_b[((t * ktn + kt) * 32 + lane) * 16];
      b.q[0] = *(const uint4*)(bp);
      b.q[1] = *(const uint4*)(bp + 8);
      acc[t] = __builtin_amdgcn_wmma_f32_16x16x32_bf16(
          false, a.v, false, b.v, (short)0, acc[t], false, false);
    }
    if (A2) {
      Frag a2;
      a2.q[0] = *(const uint4*)(a2p + kt * 32);
      a2.q[1] = *(const uint4*)(a2p + kt * 32 + 8);
#pragma unroll
      for (int t = 0; t < 4; ++t) {
        Frag b;
        const unsigned short* bp = &lds_b[panelElems + ((t * ktn + kt) * 32 + lane) * 16];
        b.q[0] = *(const uint4*)(bp);
        b.q[1] = *(const uint4*)(bp + 8);
        acc[t] = __builtin_amdgcn_wmma_f32_16x16x32_bf16(
            false, a2.v, false, b.v, (short)0, acc[t], false, false);
      }
    }
  }

  // Epilogue. C/D layout: lane<16 -> rows v (VGPR v), lane>=16 -> rows 8+v;
  // column = tile*16 + (lane&15).
#pragma unroll
  for (int t = 0; t < 4; ++t) {
    int n = (cg * 4 + t) * 16 + l15;
    float bs = 0.f;
    if (bias1) bs += bias1[n];
    if (bias2) bs += bias2[n];
#pragma unroll
    for (int v = 0; v < 8; ++v) {
      int row = rt * 16 + half * 8 + v;
      float val = acc[t][v] + bs;
      if (relu) val = fmaxf(val, 0.f);
      size_t idx = (size_t)row * N + n;
      if (resid) val += resid[idx];
      if (outF) outF[idx] = val;
      else      outB[idx] = f2bf(val);
    }
  }
}

// ---------------------------------------------------------------------------
// scores[r,h] = sum_{c in head h} k[r,c] * q[batch[e]*M+m, c]; one wave/row.
// lane covers channels lane*8..lane*8+7 => head = lane>>2; reduce over 4 lanes.
// ---------------------------------------------------------------------------
__global__ void mme_scores(const unsigned short* __restrict__ Kb,
                           const float* __restrict__ Q,
                           const int* __restrict__ batch,
                           float* __restrict__ S) {
  int lane = threadIdx.x & 31;
  int r = blockIdx.x * (blockDim.x >> 5) + (threadIdx.x >> 5);
  if (r >= R_EDGE) return;
  int e = r / M_MODES_;
  int m = r - e * M_MODES_;
  size_t qoff = ((size_t)batch[e] * M_MODES_ + m) * HID_;
  size_t koff = (size_t)r * HID_;
  int c0 = lane * 8;
  uint4 kq = *(const uint4*)(Kb + koff + c0);
  const unsigned short* kp = (const unsigned short*)&kq;
  float s = 0.f;
#pragma unroll
  for (int j = 0; j < 8; ++j) s += bf2f(kp[j]) * Q[qoff + c0 + j];
  s += __shfl_xor(s, 1);
  s += __shfl_xor(s, 2);
  if ((lane & 3) == 0) S[(size_t)r * HEADS_ + (lane >> 2)] = s;
}

// ---------------------------------------------------------------------------
// Segment softmax over edges (batch is sorted) + weighted scatter-sum of v,
// residual-added into ego in place. One 256-thread block per ego node.
// ---------------------------------------------------------------------------
__global__ void mme_attn_agg(const float* __restrict__ S,
                             const unsigned short* __restrict__ V,
                             const int* __restrict__ batch,
                             float* __restrict__ ego) {
  __shared__ float s_mx[48], s_den[48], s_al[48];
  int n = blockIdx.x;
  int t = threadIdx.x;

  int lo = 0, hi = E_EDGE;                       // lower_bound(n)
  while (lo < hi) { int mid = (lo + hi) >> 1; if (batch[mid] < n) lo = mid + 1; else hi = mid; }
  int start = lo;
  hi = E_EDGE;                                   // lower_bound(n+1)
  while (lo < hi) { int mid = (lo + hi) >> 1; if (batch[mid] < n + 1) lo = mid + 1; else hi = mid; }
  int end = lo;

  if (t < 48) {                                  // t = m*8 + h
    int m = t >> 3, h = t & 7;
    float mx = -3.0e38f;
    for (int e = start; e < end; ++e)
      mx = fmaxf(mx, S[(size_t)(e * M_MODES_ + m) * HEADS_ + h]);
    float den = 0.f;
    for (int e = start; e < end; ++e)
      den += expf(S[(size_t)(e * M_MODES_ + m) * HEADS_ + h] - mx);
    s_mx[t] = mx;
    s_den[t] = den + 1e-16f;
  }
  __syncthreads();

  float acc[6];
  int mj[6], cj[6], aj[6];
#pragma unroll
  for (int j = 0; j < 6; ++j) {                  // each thread owns 6 of M*H=1536
    int f = t + 256 * j;
    mj[j] = f >> 8;
    cj[j] = f & 255;
    aj[j] = mj[j] * 8 + (cj[j] >> 5);
    acc[j] = 0.f;
  }
  for (int e = start; e < end; ++e) {
    if (t < 48)
      s_al[t] = expf(S[(size_t)(e * M_MODES_ + (t >> 3)) * HEADS_ + (t & 7)] - s_mx[t]) / s_den[t];
    __syncthreads();
#pragma unroll
    for (int j = 0; j < 6; ++j)
      acc[j] += s_al[aj[j]] * bf2f(V[(size_t)(e * M_MODES_ + mj[j]) * HID_ + cj[j]]);
    __syncthreads();
  }
#pragma unroll
  for (int j = 0; j < 6; ++j) {
    size_t idx = (size_t)(n * M_MODES_ + mj[j]) * HID_ + cj[j];
    ego[idx] += acc[j];
  }
}

// ---------------------------------------------------------------------------
// Host-side orchestration
// ---------------------------------------------------------------------------
extern "C" void kernel_launch(void* const* d_in, const int* in_sizes, int n_in,
                              void* d_out, int out_size, void* d_ws, size_t ws_size,
                              hipStream_t stream) {
  const int*   batch = (const int*)  d_in[0];
  const float* egoF  = (const float*)d_in[1];
  const float* edgeF = (const float*)d_in[2];
  const float* nodeF = (const float*)d_in[3];
  const float* Wk  = (const float*)d_in[4];   const float* bk  = (const float*)d_in[5];
  const float* Wq  = (const float*)d_in[6];   const float* bq  = (const float*)d_in[7];
  const float* Wv1 = (const float*)d_in[8];   const float* bv1 = (const float*)d_in[9];
  const float* Wv2 = (const float*)d_in[10];  const float* bv2 = (const float*)d_in[11];
  const float* ln1w= (const float*)d_in[12];  const float* ln1b= (const float*)d_in[13];
  const float* ln2w= (const float*)d_in[14];  const float* ln2b= (const float*)d_in[15];
  const float* W1  = (const float*)d_in[16];  const float* b1  = (const float*)d_in[17];
  const float* W2  = (const float*)d_in[18];  const float* b2  = (const float*)d_in[19];
  const float* Wd  = (const float*)d_in[20];  const float* bd  = (const float*)d_in[21];

  char* wp = (char*)d_ws;
  auto alloc = [&](size_t bytes) -> char* {
    char* p = wp; wp += (bytes + 255) & ~(size_t)255; return p;
  };
  unsigned short* edge_bf = (unsigned short*)alloc((size_t)R_EDGE * HH_  * 2);
  unsigned short* node_bf = (unsigned short*)alloc((size_t)R_EDGE * HH_  * 2);
  unsigned short* k_bf    = (unsigned short*)alloc((size_t)R_EDGE * HID_ * 2);
  unsigned short* v_bf    = (unsigned short*)alloc((size_t)R_EDGE * HID_ * 2);
  float*          scores  = (float*)         alloc((size_t)R_EDGE * HEADS_ * 4);
  float*          q_f     = (float*)         alloc((size_t)R_EGO  * HID_ * 4);
  unsigned short* xn_bf   = (unsigned short*)alloc((size_t)R_EGO  * HID_ * 2);
  unsigned short* t1_bf   = (unsigned short*)alloc((size_t)R_EGO  * HID_ * 2);
  unsigned short* t2_bf   = (unsigned short*)alloc((size_t)R_EGO  * HID_ * 2);
  float*          ego_buf = (float*)         alloc((size_t)R_EGO  * HID_ * 4);
  unsigned short* Wk_p  = (unsigned short*)alloc((size_t)L_ * HH_  * HID_ * 2);
  unsigned short* Wv1_p = (unsigned short*)alloc((size_t)L_ * HH_  * HID_ * 2);
  unsigned short* Wv2_p = (unsigned short*)alloc((size_t)L_ * HH_  * HID_ * 2);
  unsigned short* Wq_p  = (unsigned short*)alloc((size_t)L_ * HID_ * HID_ * 2);
  unsigned short* W1_p  = (unsigned short*)alloc((size_t)L_ * HID_ * HID_ * 2);
  unsigned short* W2_p  = (unsigned short*)alloc((size_t)L_ * HID_ * HID_ * 2);
  unsigned short* Wd_p  = (unsigned short*)alloc((size_t)L_ * HID_ * HID_ * 2);

  auto gemm = [&](const unsigned short* A1, const unsigned short* B1,
                  const unsigned short* A2, const unsigned short* B2,
                  const float* bias1, const float* bias2, const float* resid,
                  float* outF, unsigned short* outB, int R, int K, int N, int relu) {
    int blocks = (R / 128) * (N / 64);   // 8 row-tiles x 1 column group each
    mme_gemm_wmma<<<blocks, 256, 0, stream>>>(A1, B1, A2, B2, bias1, bias2,
                                              resid, outF, outB, R, K, N, relu);
  };

  // One-time: features & weights -> bf16 (re-done every call for determinism).
  mme_cvt_bf16<<<2048, 256, 0, stream>>>(edgeF, edge_bf, (size_t)R_EDGE * HH_);
  mme_cvt_bf16<<<2048, 256, 0, stream>>>(nodeF, node_bf, (size_t)R_EDGE * HH_);
  for (int l = 0; l < L_; ++l) {
    size_t o128 = (size_t)l * HH_ * HID_, o256 = (size_t)l * HID_ * HID_;
    int g128 = (HH_  / 32) * (HID_ / 16);
    int g256 = (HID_ / 32) * (HID_ / 16);
    mme_pack_b<<<g128, 32, 0, stream>>>(Wk  + o128, Wk_p  + o128, HH_,  HID_);
    mme_pack_b<<<g128, 32, 0, stream>>>(Wv1 + o128, Wv1_p + o128, HH_,  HID_);
    mme_pack_b<<<g128, 32, 0, stream>>>(Wv2 + o128, Wv2_p + o128, HH_,  HID_);
    mme_pack_b<<<g256, 32, 0, stream>>>(Wq  + o256, Wq_p  + o256, HID_, HID_);
    mme_pack_b<<<g256, 32, 0, stream>>>(W1  + o256, W1_p  + o256, HID_, HID_);
    mme_pack_b<<<g256, 32, 0, stream>>>(W2  + o256, W2_p  + o256, HID_, HID_);
    mme_pack_b<<<g256, 32, 0, stream>>>(Wd  + o256, Wd_p  + o256, HID_, HID_);
  }
  hipMemcpyAsync(ego_buf, egoF, (size_t)R_EGO * HID_ * 4,
                 hipMemcpyDeviceToDevice, stream);

  for (int l = 0; l < L_; ++l) {
    size_t o128 = (size_t)l * HH_ * HID_, o256 = (size_t)l * HID_ * HID_;
    size_t bo   = (size_t)l * HID_;
    // --- attention block ---
    mme_layernorm<<<R_EGO / 8, 256, 0, stream>>>(ego_buf, ln1w + bo, ln1b + bo, xn_bf, R_EGO);
    gemm(xn_bf, Wq_p + o256, 0, 0, bq + bo, 0, 0, q_f, 0, R_EGO, HID_, HID_, 0);
    gemm(edge_bf, Wk_p + o128, 0, 0, bk + bo, 0, 0, 0, k_bf, R_EDGE, HH_, HID_, 0);
    gemm(edge_bf, Wv1_p + o128, node_bf, Wv2_p + o128,
         bv1 + bo, bv2 + bo, 0, 0, v_bf, R_EDGE, HH_, HID_, 0);
    mme_scores<<<R_EDGE / 8, 256, 0, stream>>>(k_bf, q_f, batch, scores);
    mme_attn_agg<<<N_EGO_, 256, 0, stream>>>(scores, v_bf, batch, ego_buf);
    // --- ff block ---
    mme_layernorm<<<R_EGO / 8, 256, 0, stream>>>(ego_buf, ln2w + bo, ln2b + bo, xn_bf, R_EGO);
    gemm(xn_bf, W1_p + o256, 0, 0, b1 + bo, 0, 0, 0, t1_bf, R_EGO, HID_, HID_, 1);
    gemm(t1_bf, Wd_p + o256, 0, 0, bd + bo, 0, 0, 0, t2_bf, R_EGO, HID_, HID_, 0);
    gemm(t2_bf, W2_p + o256, 0, 0, b2 + bo, 0, ego_buf, ego_buf, 0, R_EGO, HID_, HID_, 0);
  }

  hipMemcpyAsync(d_out, ego_buf, (size_t)R_EGO * HID_ * 4,
                 hipMemcpyDeviceToDevice, stream);
}
